// TunedSparseAttention1D_41979010351635
// MI455X (gfx1250) — compile-verified, matrix operand order checked by reading, not run on hardware
//
#include <hip/hip_runtime.h>
#include <hip/hip_bf16.h>

typedef _Float16 v16h __attribute__((ext_vector_type(16)));
typedef _Float16 h8   __attribute__((ext_vector_type(8)));
typedef float    v8f  __attribute__((ext_vector_type(8)));

#define TOK   32768   // B*N tokens
#define DIM   1024
#define NHEAD 16
#define HD    64
#define WIN   128

// ---------------------------------------------------------------------------
// 16-bit A/B fragment loader for V_WMMA_F32_16X16X32_F16.
// Per CDNA5 ISA 7.12.2: lanes 0-15 hold row (lane&15), K = {0..7} in v0..3 and
// {16..23} in v4..7; lanes 16-31 hold the same rows, K = {8..15} and {24..31}.
// Both halves are two contiguous 8-half (16 B) chunks -> two b128 loads.
// ---------------------------------------------------------------------------
__device__ __forceinline__ v16h load_frag(const _Float16* base, int ld) {
  const int lane = threadIdx.x & 31;
  const _Float16* p = base + (size_t)(lane & 15) * ld + ((lane >> 4) << 3);
  h8 lo = *(const h8*)(p);
  h8 hi = *(const h8*)(p + 16);
  return __builtin_shufflevector(lo, hi, 0,1,2,3,4,5,6,7,8,9,10,11,12,13,14,15);
}

__device__ __forceinline__ float gelu_exact(float x) {
  return 0.5f * x * (1.0f + erff(x * 0.70710678118654752440f));
}

// ---------------------------------------------------------------------------
// CDNA5 async global->LDS copy (GLOBAL_LOAD_ASYNC_TO_LDS_B128, ASYNCcnt).
// LDS operand is the byte offset within the wave's LDS allocation; for a
// generic pointer into __shared__, that is addr[31:0] (flat aperture rule).
// ---------------------------------------------------------------------------
__device__ __forceinline__ void async_ld_b128(void* lds_dst, const void* gsrc) {
  unsigned int loff = (unsigned int)(size_t)lds_dst;
  asm volatile("global_load_async_to_lds_b128 %0, %1, off"
               :: "v"(loff), "v"(gsrc) : "memory");
}

__device__ __forceinline__ void wait_async0() {
  asm volatile("s_wait_asynccnt 0" ::: "memory");
}

// ---------------------------------------------------------------------------
// f32 -> f16 weight conversion
// ---------------------------------------------------------------------------
__global__ void cvt_kernel(const float* __restrict__ src,
                           _Float16* __restrict__ dst, int n) {
  int i = blockIdx.x * 256 + threadIdx.x;
  if (i < n) dst[i] = (_Float16)src[i];
}

// ---------------------------------------------------------------------------
// LayerNorm: one wave (32 lanes) per token, 32 channels per lane.
// ---------------------------------------------------------------------------
__global__ __launch_bounds__(256) void ln_kernel(
    const float* __restrict__ x, const float* __restrict__ w,
    const float* __restrict__ b, _Float16* __restrict__ out, int ntok) {
  const int lane = threadIdx.x & 31;
  const int wave = threadIdx.x >> 5;
  const size_t token = (size_t)blockIdx.x * 8 + wave;
  if (token >= (size_t)ntok) return;
  const float* row = x + token * DIM;
  float vals[32];
  float s = 0.f;
#pragma unroll
  for (int j = 0; j < 32; ++j) { vals[j] = row[lane + j * 32]; s += vals[j]; }
#pragma unroll
  for (int off = 16; off >= 1; off >>= 1) s += __shfl_xor(s, off, 32);
  const float mu = s * (1.0f / DIM);
  float var = 0.f;
#pragma unroll
  for (int j = 0; j < 32; ++j) { float d = vals[j] - mu; var += d * d; }
#pragma unroll
  for (int off = 16; off >= 1; off >>= 1) var += __shfl_xor(var, off, 32);
  const float rstd = rsqrtf(var * (1.0f / DIM) + 1e-5f);
  _Float16* orow = out + token * DIM;
#pragma unroll
  for (int j = 0; j < 32; ++j) {
    int c = lane + j * 32;
    orow[c] = (_Float16)((vals[j] - mu) * rstd * w[c] + b[c]);
  }
}

// ---------------------------------------------------------------------------
// Tiled WMMA GEMM: Y(M,N) = X(M,K) @ W(N,K)^T + bias, with mode epilogues.
// Block = 256 threads (8 waves, 4x2), tile = 128x128, K-step = 32.
// Double-buffered LDS staged via async global->LDS (ASYNCcnt pipeline):
//   wait_async(prev) -> barrier -> issue async(next) -> 8 WMMAs on current.
// MODE 0: scatter f16 into q/k/v laid out (B,H,N,HD)
// MODE 1: f32 out = acc + bias + residual        (attention proj)
// MODE 2: f16 out = gelu(acc + bias)             (FC1)
// MODE 3: f32 out = acc + bias + residual        (FC2 -> d_out)
// ---------------------------------------------------------------------------
#define BK 32
#define LDA 40  // halfs pitch (32 + 8 pad) -> 80B rows, 16B-aligned chunks

template <int MODE>
__global__ __launch_bounds__(256) void gemm_kernel(
    const _Float16* __restrict__ X, const _Float16* __restrict__ Wt,
    const float* __restrict__ bias, const float* __restrict__ residual,
    void* __restrict__ outp, int M, int N, int K) {
  __shared__ _Float16 As[2][128 * LDA];
  __shared__ _Float16 Bs[2][128 * LDA];

  const int tid  = threadIdx.x;
  const int lane = tid & 31, wave = tid >> 5;
  const int half = lane >> 4, lr = lane & 15;
  const int wm = wave >> 1;  // 0..3 -> 32-row strip
  const int wn = wave & 1;   // 0..1 -> 64-col strip
  const int m0 = blockIdx.y * 128;
  const int n0 = blockIdx.x * 128;

  // stage one 128x32 K-slice of A and B into LDS buffer `buf` (async)
  auto stage = [&](int buf, int k0) {
#pragma unroll
    for (int p = 0; p < 2; ++p) {
      int idx = tid + p * 256;              // 0..511
      int r = idx >> 2, c = (idx & 3) << 3; // 128 rows x 4 chunks of 8 halfs
      async_ld_b128(&As[buf][r * LDA + c], &X[(size_t)(m0 + r) * K + k0 + c]);
      async_ld_b128(&Bs[buf][r * LDA + c], &Wt[(size_t)(n0 + r) * K + k0 + c]);
    }
  };

  v8f acc[2][4];
#pragma unroll
  for (int i = 0; i < 2; ++i)
#pragma unroll
    for (int j = 0; j < 4; ++j) acc[i][j] = {};

  int cur = 0;
  stage(0, 0);
  for (int k0 = 0; k0 < K; k0 += BK) {
    wait_async0();        // my async writes done (per-wave ASYNCcnt)
    __syncthreads();      // publish all waves' writes; prev reads finished
    if (k0 + BK < K) stage(cur ^ 1, k0 + BK);

    v16h a[2], b[4];
#pragma unroll
    for (int i = 0; i < 2; ++i)
      a[i] = load_frag(&As[cur][(wm * 32 + i * 16) * LDA], LDA);
#pragma unroll
    for (int j = 0; j < 4; ++j)
      b[j] = load_frag(&Bs[cur][(wn * 64 + j * 16) * LDA], LDA);
#pragma unroll
    for (int i = 0; i < 2; ++i)
#pragma unroll
      for (int j = 0; j < 4; ++j)
        acc[i][j] = __builtin_amdgcn_wmma_f32_16x16x32_f16(
            false, a[i], false, b[j], (short)0, acc[i][j], false, false);
    cur ^= 1;
  }

  // Epilogue. C/D layout: vreg r holds row (r + 8*half), column (lane&15).
#pragma unroll
  for (int i = 0; i < 2; ++i) {
#pragma unroll
    for (int j = 0; j < 4; ++j) {
      const int mt = m0 + wm * 32 + i * 16;
      const int nt = n0 + wn * 64 + j * 16;
      const int col = nt + lr;
      const float bv = bias[col];
#pragma unroll
      for (int r = 0; r < 8; ++r) {
        const int row = mt + r + 8 * half;
        const float vv = acc[i][j][r] + bv;
        if (MODE == 0) {
          // col in [0,3072): s = q/k/v, head h, dim d; row = b*8192 + n
          const int s  = col >> 10;
          const int rm = col & 1023;
          const int hh = rm >> 6, dd = rm & 63;
          const size_t bb = (size_t)(row >> 13), nn = (size_t)(row & 8191);
          _Float16* o = (_Float16*)outp;
          o[((size_t)s << 25) + ((bb * NHEAD + hh) * 8192 + nn) * HD + dd] =
              (_Float16)vv;
        } else if (MODE == 2) {
          _Float16* o = (_Float16*)outp;
          o[(size_t)row * N + col] = (_Float16)gelu_exact(vv);
        } else {  // MODE 1 / 3: f32 + residual
          float* o = (float*)outp;
          const size_t idx = (size_t)row * N + col;
          o[idx] = vv + residual[idx];
        }
      }
    }
  }
}

// ---------------------------------------------------------------------------
// Windowed attention: one block per (b, head, window). 8 waves; wave w owns
// query rows [16w, 16w+16). S = Q K^T (WMMA), register softmax with 16-lane
// shfl_xor reductions, P staged to LDS (f16), O = P V (WMMA, V transposed
// through LDS so its K-dim is contiguous for B fragments).
// ---------------------------------------------------------------------------
__global__ __launch_bounds__(256) void attn_kernel(
    const _Float16* __restrict__ q, const _Float16* __restrict__ k,
    const _Float16* __restrict__ v, _Float16* __restrict__ o) {
  __shared__ _Float16 Vt[64 * 136];   // (d, k) pitch 136
  __shared__ _Float16 Ps[128 * 136];  // (row, k) pitch 136

  const int bid = blockIdx.x;          // ((b*16+h)*64 + win)
  const int win = bid & 63;
  const int bh  = bid >> 6;
  const size_t base = ((size_t)bh * 8192 + (size_t)win * WIN) * HD;
  const _Float16* Q = q + base;  // 128 x 64 row-major
  const _Float16* K = k + base;
  const _Float16* V = v + base;

  for (int idx = threadIdx.x; idx < WIN * HD; idx += 256) {
    int kk = idx >> 6, dd = idx & 63;
    Vt[dd * 136 + kk] = V[idx];
  }
  __syncthreads();

  const int lane = threadIdx.x & 31, wave = threadIdx.x >> 5;
  const int half = lane >> 4, lr = lane & 15;
  const int r0 = wave * 16;

  // ---- S = Q K^T * 1/sqrt(64), strip of 16 rows x 128 cols (8 tiles) ----
  v8f s[8];
#pragma unroll
  for (int t = 0; t < 8; ++t) s[t] = {};
#pragma unroll
  for (int kk = 0; kk < HD; kk += 32) {
    v16h aq = load_frag(Q + (size_t)r0 * HD + kk, HD);
#pragma unroll
    for (int t = 0; t < 8; ++t) {
      v16h bk = load_frag(K + (size_t)t * 16 * HD + kk, HD);
      s[t] = __builtin_amdgcn_wmma_f32_16x16x32_f16(
          false, aq, false, bk, (short)0, s[t], false, false);
    }
  }

  // ---- softmax: row (r0 + i + 8*half) lives in lanes of one 16-lane half ----
#pragma unroll
  for (int i = 0; i < 8; ++i) {
    float m = -3.0e38f;
#pragma unroll
    for (int t = 0; t < 8; ++t) {
      float vv = s[t][i] * 0.125f;
      s[t][i] = vv;
      m = fmaxf(m, vv);
    }
#pragma unroll
    for (int off = 1; off < 16; off <<= 1) m = fmaxf(m, __shfl_xor(m, off, 32));
    float sum = 0.f;
#pragma unroll
    for (int t = 0; t < 8; ++t) {
      float e = __expf(s[t][i] - m);
      s[t][i] = e;
      sum += e;
    }
#pragma unroll
    for (int off = 1; off < 16; off <<= 1) sum += __shfl_xor(sum, off, 32);
    const float inv = 1.0f / sum;
    const int row = r0 + i + 8 * half;
#pragma unroll
    for (int t = 0; t < 8; ++t)
      Ps[row * 136 + t * 16 + lr] = (_Float16)(s[t][i] * inv);
  }
  // Same-wave LDS store->load is in-order; each wave reads only its own strip.

  // ---- O = P V : 16 rows x 64 cols (4 tiles), K = 128 ----
  v8f oa[4];
#pragma unroll
  for (int dt = 0; dt < 4; ++dt) oa[dt] = {};
#pragma unroll
  for (int kk = 0; kk < WIN; kk += 32) {
    v16h ap = load_frag(&Ps[r0 * 136 + kk], 136);
#pragma unroll
    for (int dt = 0; dt < 4; ++dt) {
      v16h bv = load_frag(&Vt[dt * 16 * 136 + kk], 136);
      oa[dt] = __builtin_amdgcn_wmma_f32_16x16x32_f16(
          false, ap, false, bv, (short)0, oa[dt], false, false);
    }
  }

  // scatter to (B, N, C) with heads concatenated: col = h*64 + d
  const int b = bh >> 4, h = bh & 15;
#pragma unroll
  for (int dt = 0; dt < 4; ++dt) {
#pragma unroll
    for (int r = 0; r < 8; ++r) {
      const int n = win * WIN + r0 + r + 8 * half;
      const size_t token = (size_t)b * 8192 + n;
      const int col = h * HD + dt * 16 + lr;
      o[token * DIM + col] = (_Float16)oa[dt][r];
    }
  }
}

// ---------------------------------------------------------------------------
// Host-side orchestration
// ---------------------------------------------------------------------------
extern "C" void kernel_launch(void* const* d_in, const int* in_sizes, int n_in,
                              void* d_out, int out_size, void* d_ws,
                              size_t ws_size, hipStream_t stream) {
  const float* x      = (const float*)d_in[0];
  const float* ln1_w  = (const float*)d_in[1];
  const float* ln1_b  = (const float*)d_in[2];
  const float* qkv_w  = (const float*)d_in[3];
  const float* qkv_b  = (const float*)d_in[4];
  const float* proj_w = (const float*)d_in[5];
  const float* proj_b = (const float*)d_in[6];
  const float* ln2_w  = (const float*)d_in[7];
  const float* ln2_b  = (const float*)d_in[8];
  const float* fc1_w  = (const float*)d_in[9];
  const float* fc1_b  = (const float*)d_in[10];
  const float* fc2_w  = (const float*)d_in[11];
  const float* fc2_b  = (const float*)d_in[12];
  float* out = (float*)d_out;

  char* ws = (char*)d_ws;
  const size_t MB = 1024ull * 1024ull;
  // [0,16MB): f16 weights
  _Float16* qkv_w16  = (_Float16*)ws;
  _Float16* proj_w16 = qkv_w16 + 3u * DIM * DIM;
  _Float16* fc1_w16  = proj_w16 + (size_t)DIM * DIM;
  _Float16* fc2_w16  = fc1_w16 + 2u * DIM * DIM;
  // [16,80): LN output (xn, reused as h)      64 MB f16
  _Float16* xn = (_Float16*)(ws + 16 * MB);
  // [80,272): q/k/v each (B,H,N,HD)           3 x 64 MB f16; act reuses front
  _Float16* qbuf = (_Float16*)(ws + 80 * MB);
  _Float16* kbuf = qbuf + ((size_t)1 << 25);
  _Float16* vbuf = kbuf + ((size_t)1 << 25);
  _Float16* act  = qbuf;  // 128 MB, reused after attention consumes q/k/v
  // [272,336): attention output               64 MB f16
  _Float16* attn_out = (_Float16*)(ws + 272 * MB);
  // [336,464): x1 = x + attn (f32 residual)   128 MB
  float* x1 = (float*)(ws + 336 * MB);

  // 1) weight casts
  cvt_kernel<<<(3 * DIM * DIM + 255) / 256, 256, 0, stream>>>(qkv_w, qkv_w16, 3 * DIM * DIM);
  cvt_kernel<<<(DIM * DIM + 255) / 256, 256, 0, stream>>>(proj_w, proj_w16, DIM * DIM);
  cvt_kernel<<<(2 * DIM * DIM + 255) / 256, 256, 0, stream>>>(fc1_w, fc1_w16, 2 * DIM * DIM);
  cvt_kernel<<<(2 * DIM * DIM + 255) / 256, 256, 0, stream>>>(fc2_w, fc2_w16, 2 * DIM * DIM);

  // 2) LN1
  ln_kernel<<<TOK / 8, 256, 0, stream>>>(x, ln1_w, ln1_b, xn, TOK);

  // 3) QKV GEMM (32768 x 3072 x 1024), scatter into (B,H,N,HD) q/k/v
  gemm_kernel<0><<<dim3(3072 / 128, TOK / 128), 256, 0, stream>>>(
      xn, qkv_w16, qkv_b, nullptr, (void*)qbuf, TOK, 3072, DIM);

  // 4) windowed attention: 4*16*64 blocks
  attn_kernel<<<4 * NHEAD * 64, 256, 0, stream>>>(qbuf, kbuf, vbuf, attn_out);

  // 5) proj GEMM + residual -> x1 (f32)
  gemm_kernel<1><<<dim3(DIM / 128, TOK / 128), 256, 0, stream>>>(
      attn_out, proj_w16, proj_b, x, (void*)x1, TOK, DIM, DIM);

  // 6) LN2
  ln_kernel<<<TOK / 8, 256, 0, stream>>>(x1, ln2_w, ln2_b, xn, TOK);

  // 7) FC1 GEMM + exact GELU (32768 x 2048 x 1024)
  gemm_kernel<2><<<dim3(2048 / 128, TOK / 128), 256, 0, stream>>>(
      xn, fc1_w16, fc1_b, nullptr, (void*)act, TOK, 2048, DIM);

  // 8) FC2 GEMM + residual -> d_out (32768 x 1024 x 2048)
  gemm_kernel<3><<<dim3(DIM / 128, TOK / 128), 256, 0, stream>>>(
      act, fc2_w16, fc2_b, x1, (void*)out, TOK, DIM, 2048);
}
